// DynamicAdjGenerator_39410619908371
// MI455X (gfx1250) — matrix operand order repaired and checked by reading, not compiled, not run on hardware
//
#include <hip/hip_runtime.h>

#define TT 64
#define BB 512
#define DD 768
#define HH 12
#define HDD 64
#define D3 2304
#define TBM (TT * BB)

typedef __attribute__((ext_vector_type(16))) __bf16 bf16x16;
typedef __attribute__((ext_vector_type(8)))  __bf16 bf16x8;
typedef __attribute__((ext_vector_type(8)))  float  f32x8;
typedef __attribute__((ext_vector_type(4)))  unsigned int u32x4;
typedef __attribute__((ext_vector_type(8)))  int    i32x8;
typedef __attribute__((ext_vector_type(4)))  int    i32x4;

#if defined(__has_builtin)
#if __has_builtin(__builtin_amdgcn_permlane16)
#define HAVE_PERMLANE16 1
#endif
#if __has_builtin(__builtin_amdgcn_tensor_load_to_lds) && \
    __has_builtin(__builtin_amdgcn_s_wait_tensorcnt)
#define HAVE_TDM 1
#endif
#endif

// ---------- helpers ----------

__device__ inline __bf16 bfc(float f) { return (__bf16)f; }  // native RNE cvt

__device__ inline bf16x16 cat16(bf16x8 lo, bf16x8 hi) {
  bf16x16 a;
#pragma unroll
  for (int i = 0; i < 8; ++i) { a[i] = lo[i]; a[i + 8] = hi[i]; }
  return a;
}

__device__ inline f32x8 wmma_bf16(bf16x16 a, bf16x16 b, f32x8 c) {
  return __builtin_amdgcn_wmma_f32_16x16x32_bf16(false, a, false, b,
                                                 (short)0, c, false, false);
}

// xor-shuffle within each 16-lane half. v_permlane16_b32 (VALU) when available,
// else ds_bpermute via __shfl_xor. Select tables: nibble k = source lane for lane k.
template <unsigned SLO, unsigned SHI, int MSK>
__device__ inline float permx(float v) {
#ifdef HAVE_PERMLANE16
  unsigned i = __builtin_bit_cast(unsigned, v);
  unsigned r = __builtin_amdgcn_permlane16(i, i, SLO, SHI, false, false);
  return __builtin_bit_cast(float, r);
#else
  return __shfl_xor(v, MSK, 32);
#endif
}

__device__ inline float halfred_max(float v) {
  v = fmaxf(v, permx<0x67452301u, 0xEFCDAB89u, 1>(v));  // xor 1
  v = fmaxf(v, permx<0x54761032u, 0xDCFE98BAu, 2>(v));  // xor 2
  v = fmaxf(v, permx<0x32107654u, 0xBA98FEDCu, 4>(v));  // xor 4
  v = fmaxf(v, permx<0xFEDCBA98u, 0x76543210u, 8>(v));  // xor 8
  return v;
}

__device__ inline float halfred_sum(float v) {
  v += permx<0x67452301u, 0xEFCDAB89u, 1>(v);
  v += permx<0x54761032u, 0xDCFE98BAu, 2>(v);
  v += permx<0x32107654u, 0xBA98FEDCu, 4>(v);
  v += permx<0xFEDCBA98u, 0x76543210u, 8>(v);
  return v;
}

// ---------- kernel 1: f32 -> bf16 convert ----------

__global__ void cvt_bf16_kernel(const float* __restrict__ src,
                                __bf16* __restrict__ dst, int n) {
  int i = blockIdx.x * blockDim.x + threadIdx.x;
  int stride = gridDim.x * blockDim.x;
  for (; i < n; i += stride) dst[i] = bfc(src[i]);
}

// ---------- kernel 2/4: C[m,n] = sum_k X[m,k]*W[n,k] + bias[n] ----------

template <int NTOT, bool OUT_BF16>
__global__ __launch_bounds__(256) void gemm_xwT_kernel(
    const __bf16* __restrict__ X, const __bf16* __restrict__ W,
    const float* __restrict__ bias, void* __restrict__ out, int Kdim) {
  __shared__ __bf16 Xs[64 * 40];  // 80B row stride: 16B aligned, bank-spread
  __shared__ __bf16 Ws[32 * 40];

  const int m0 = blockIdx.x * 64;
  const int n0 = blockIdx.y * 32;
  const int tid = threadIdx.x;
  const int wid = tid >> 5;
  const int lane = tid & 31;
  const int half = lane >> 4;
  const int l16 = lane & 15;
  const int wm = wid & 3;
  const int wn = wid >> 2;
  const int sr = tid >> 2, sc = tid & 3;

  f32x8 acc = {0.f, 0.f, 0.f, 0.f, 0.f, 0.f, 0.f, 0.f};

  for (int k0 = 0; k0 < Kdim; k0 += 32) {
    *(bf16x8*)&Xs[sr * 40 + sc * 8] =
        *(const bf16x8*)&X[(size_t)(m0 + sr) * Kdim + k0 + sc * 8];
    if (tid < 128)
      *(bf16x8*)&Ws[sr * 40 + sc * 8] =
          *(const bf16x8*)&W[(size_t)(n0 + sr) * Kdim + k0 + sc * 8];
    if (k0 + 32 < Kdim) {  // global_prefetch_b8 for next K tile
      __builtin_prefetch(&X[(size_t)(m0 + sr) * Kdim + k0 + 32 + sc * 8], 0, 1);
      if (tid < 128)
        __builtin_prefetch(&W[(size_t)(n0 + sr) * Kdim + k0 + 32 + sc * 8], 0, 1);
    }
    __syncthreads();
    bf16x8 alo = *(const bf16x8*)&Xs[(wm * 16 + l16) * 40 + half * 8];
    bf16x8 ahi = *(const bf16x8*)&Xs[(wm * 16 + l16) * 40 + half * 8 + 16];
    bf16x8 blo = *(const bf16x8*)&Ws[(wn * 16 + l16) * 40 + half * 8];
    bf16x8 bhi = *(const bf16x8*)&Ws[(wn * 16 + l16) * 40 + half * 8 + 16];
    acc = wmma_bf16(cat16(alo, ahi), cat16(blo, bhi), acc);
    __syncthreads();
  }

  const int col = n0 + wn * 16 + l16;
  const float bv = bias[col];
#pragma unroll
  for (int vr = 0; vr < 8; ++vr) {
    int row = m0 + wm * 16 + vr + half * 8;
    float v = acc[vr] + bv;
    if (OUT_BF16)
      ((__bf16*)out)[(size_t)row * NTOT + col] = bfc(v);
    else
      ((float*)out)[(size_t)row * NTOT + col] = v;
  }
}

// ---------- kernel 3: flash attention per (t, h) ----------
// LDS: K[512][72] bf16 (144B rows via TDM pad), Vt[64][520] bf16, P[8][16][40] bf16

#define KLS 72
#define VTS 520
#define PLS 40
#define SM_K  (512 * KLS * 2)
#define SM_VT (64 * VTS * 2)
#define SM_P  (8 * 16 * PLS * 2)
#define ATTN_SMEM (SM_K + SM_VT + SM_P)
#define SCL 0.125f  // 1/sqrt(64)

__global__ __launch_bounds__(256) void attn_kernel(
    const __bf16* __restrict__ qkv, __bf16* __restrict__ ctx) {
  extern __shared__ char smem[];
  __bf16* Kl = (__bf16*)smem;
  __bf16* Vt = (__bf16*)(smem + SM_K);
  __bf16* Pl = (__bf16*)(smem + SM_K + SM_VT);

  const int t = blockIdx.x;
  const int h = blockIdx.y;
  const int tid = threadIdx.x;
  const int wid = tid >> 5;
  const int lane = tid & 31;
  const int half = lane >> 4;
  const int l16 = lane & 15;
  const size_t rowbase = (size_t)t * BB;

#ifdef HAVE_TDM
  // Stage K via the Tensor Data Mover: 2D tile 64(elem) x 512(rows),
  // global row stride 2304 elem, LDS pad 16B per 128B row -> 144B rows (= KLS).
  if (wid == 0) {
    unsigned lds_addr = (unsigned)(size_t)(void*)Kl;
    unsigned long long ga =
        (unsigned long long)(size_t)&qkv[rowbase * D3 + DD + h * HDD];
    u32x4 g0;
    g0[0] = 1u;                                            // count=1 (valid user D#)
    g0[1] = lds_addr;                                      // lds_addr
    g0[2] = (unsigned)(ga & 0xFFFFFFFFull);                // global_addr[31:0]
    g0[3] = (unsigned)((ga >> 32) & 0x1FFFFFFull) | (2u << 30);  // addr[56:32]|type=2
    i32x8 g1;
    g1[0] = (1 << 16)      // data_size = 2 bytes
          | (1 << 20)      // pad_enable
          | (4 << 22)      // pad_interval: 32 DWORDs (=128B row)
          | (3 << 25);     // pad_amount: 4 DWORDs (=16B)
    g1[1] = (64 << 16);    // tensor_dim0 = 64  (bits 79:48)
    g1[2] = (512 << 16);   // tensor_dim1 = 512 (bits 111:80)
    g1[3] = (64 << 16);    // tile_dim0 = 64    (bits 127:112)
    g1[4] = 512;           // tile_dim1 = 512   (bits 143:128)
    g1[5] = 2304;          // tensor_dim0_stride (bits 207:160)
    g1[6] = 0;
    g1[7] = 0;
    i32x4 gz = {0, 0, 0, 0};
#if __clang_major__ >= 23
    i32x8 gz8 = {0, 0, 0, 0, 0, 0, 0, 0};
    __builtin_amdgcn_tensor_load_to_lds(g0, g1, gz, gz, gz8, 0);
#else
    __builtin_amdgcn_tensor_load_to_lds(g0, g1, gz, gz, 0);
#endif
    __builtin_amdgcn_s_wait_tensorcnt(0);
  }
#endif

  // Stage V transposed: Vt[hd][key]
#pragma unroll
  for (int i = 0; i < 16; ++i) {
    int c = tid + i * 256;  // 4096 chunks of 8 bf16
    int key = c >> 3, hc = c & 7;
#ifndef HAVE_TDM
    bf16x8 kv = *(const bf16x8*)&qkv[(rowbase + key) * D3 + DD + h * HDD + hc * 8];
    *(bf16x8*)&Kl[key * KLS + hc * 8] = kv;
#endif
    bf16x8 vv = *(const bf16x8*)&qkv[(rowbase + key) * D3 + 2 * DD + h * HDD + hc * 8];
#pragma unroll
    for (int j = 0; j < 8; ++j) Vt[(hc * 8 + j) * VTS + key] = vv[j];
  }
  __syncthreads();

  __bf16* Plw = Pl + wid * 16 * PLS;

  for (int qb = 0; qb < 4; ++qb) {
    const int q0 = qb * 128 + wid * 16;
    const __bf16* qp = &qkv[(rowbase + q0 + l16) * D3 + h * HDD];
    bf16x16 aq0 = cat16(*(const bf16x8*)&qp[half * 8],
                        *(const bf16x8*)&qp[half * 8 + 16]);
    bf16x16 aq1 = cat16(*(const bf16x8*)&qp[32 + half * 8],
                        *(const bf16x8*)&qp[32 + half * 8 + 16]);

    float m[8], l[8];
    f32x8 O[4];
#pragma unroll
    for (int i = 0; i < 8; ++i) { m[i] = -3.0e38f; l[i] = 0.f; }
#pragma unroll
    for (int n = 0; n < 4; ++n)
      O[n] = (f32x8){0.f, 0.f, 0.f, 0.f, 0.f, 0.f, 0.f, 0.f};

    for (int kb = 0; kb < BB; kb += 32) {
      f32x8 s[2];
      s[0] = (f32x8){0.f, 0.f, 0.f, 0.f, 0.f, 0.f, 0.f, 0.f};
      s[1] = s[0];
#pragma unroll
      for (int tl = 0; tl < 2; ++tl) {
        const __bf16* kp = &Kl[(kb + tl * 16 + l16) * KLS];
        s[tl] = wmma_bf16(aq0, cat16(*(const bf16x8*)&kp[half * 8],
                                     *(const bf16x8*)&kp[half * 8 + 16]), s[tl]);
        s[tl] = wmma_bf16(aq1, cat16(*(const bf16x8*)&kp[32 + half * 8],
                                     *(const bf16x8*)&kp[32 + half * 8 + 16]), s[tl]);
      }
      // Online softmax; row r lives in element r on each 16-lane half.
#pragma unroll
      for (int vr = 0; vr < 8; ++vr) {
        float v = halfred_max(fmaxf(s[0][vr], s[1][vr]));
        float mn = fmaxf(m[vr], v);
        float f = __expf(SCL * (m[vr] - mn));
        m[vr] = mn;
        l[vr] *= f;
#pragma unroll
        for (int n = 0; n < 4; ++n) O[n][vr] *= f;
        float p0 = __expf(SCL * (s[0][vr] - mn));
        float p1 = __expf(SCL * (s[1][vr] - mn));
        s[0][vr] = p0; s[1][vr] = p1;
        l[vr] += halfred_sum(p0 + p1);
      }
      // Re-layout P: D layout -> A layout via per-wave LDS (same-wave DS order).
#pragma unroll
      for (int vr = 0; vr < 8; ++vr) {
        int row = vr + half * 8;
        Plw[row * PLS + l16] = bfc(s[0][vr]);
        Plw[row * PLS + 16 + l16] = bfc(s[1][vr]);
      }
      bf16x16 ap = cat16(*(const bf16x8*)&Plw[l16 * PLS + half * 8],
                         *(const bf16x8*)&Plw[l16 * PLS + half * 8 + 16]);
#pragma unroll
      for (int n = 0; n < 4; ++n) {
        const __bf16* vp = &Vt[(n * 16 + l16) * VTS + kb + half * 8];
        O[n] = wmma_bf16(ap, cat16(*(const bf16x8*)vp, *(const bf16x8*)(vp + 16)), O[n]);
      }
    }
#pragma unroll
    for (int n = 0; n < 4; ++n)
#pragma unroll
      for (int vr = 0; vr < 8; ++vr) {
        int row = q0 + vr + half * 8;
        ctx[(rowbase + row) * DD + h * HDD + n * 16 + l16] = bfc(O[n][vr] / l[vr]);
      }
  }
}

// ---------- kernel 5: adjacency adj[t,j] = (1/B) * (sum_i xn_i) . xn_j ----------

__global__ __launch_bounds__(256) void adj_kernel(const float* __restrict__ outm,
                                                  float* __restrict__ adj) {
  __shared__ float ninv[BB];
  __shared__ float mvec[DD];
  const int t = blockIdx.x;
  const int tid = threadIdx.x;
  const size_t base = (size_t)t * BB * DD;

  for (int r = tid; r < BB; r += 256) {
    const float* p = outm + base + (size_t)r * DD;
    float s = 0.f;
    for (int d = 0; d < DD; ++d) { float v = p[d]; s += v * v; }
    ninv[r] = 1.f / fmaxf(sqrtf(s), 1e-8f);
  }
  __syncthreads();
  for (int d = tid; d < DD; d += 256) {
    float s = 0.f;
    for (int b = 0; b < BB; ++b) s += outm[base + (size_t)b * DD + d] * ninv[b];
    mvec[d] = s;
  }
  __syncthreads();
  for (int r = tid; r < BB; r += 256) {
    const float* p = outm + base + (size_t)r * DD;
    float s = 0.f;
    for (int d = 0; d < DD; ++d) s += mvec[d] * p[d];
    adj[t * BB + r] = s * ninv[r] * (1.f / (float)BB);
  }
}

// ---------- launch ----------

extern "C" void kernel_launch(void* const* d_in, const int* in_sizes, int n_in,
                              void* d_out, int out_size, void* d_ws, size_t ws_size,
                              hipStream_t stream) {
  const float* node  = (const float*)d_in[0];  // (T,B,D)
  const float* w_in  = (const float*)d_in[1];  // (3D,D)
  const float* b_in  = (const float*)d_in[2];  // (3D)
  const float* w_out = (const float*)d_in[3];  // (D,D)
  const float* b_out = (const float*)d_in[4];  // (D)

  char* ws = (char*)d_ws;
  const size_t offX  = 0;         // X bf16 (later CTX bf16), 50,331,648 B
  const size_t offWq = 50331648;  // Wqkv bf16, 3,538,944 B
  const size_t offWo = 53870592;  // Wout bf16, 1,179,648 B
  const size_t offQK = 55050240;  // QKV bf16 (later OUT f32), 150,994,944 B
  __bf16* Xbf = (__bf16*)(ws + offX);
  __bf16* Wq  = (__bf16*)(ws + offWq);
  __bf16* Wo  = (__bf16*)(ws + offWo);
  __bf16* QKV = (__bf16*)(ws + offQK);
  float*  OUT = (float*)(ws + offQK);

  cvt_bf16_kernel<<<4096, 256, 0, stream>>>(node, Xbf, TBM * DD);
  cvt_bf16_kernel<<<1024, 256, 0, stream>>>(w_in, Wq, D3 * DD);
  cvt_bf16_kernel<<<512, 256, 0, stream>>>(w_out, Wo, DD * DD);

  gemm_xwT_kernel<D3, true><<<dim3(TBM / 64, D3 / 32), 256, 0, stream>>>(
      Xbf, Wq, b_in, (void*)QKV, DD);

  attn_kernel<<<dim3(TT, HH), 256, ATTN_SMEM, stream>>>(QKV, Xbf);

  gemm_xwT_kernel<DD, false><<<dim3(TBM / 64, DD / 32), 256, 0, stream>>>(
      Xbf, Wo, b_out, (void*)OUT, DD);

  adj_kernel<<<TT, 256, 0, stream>>>(OUT, (float*)d_out);
}
